// GCNModel_15470472200798
// MI455X (gfx1250) — matrix operand order repaired and checked by reading, not compiled, not run on hardware
//
#include <hip/hip_runtime.h>

typedef __attribute__((ext_vector_type(16))) _Float16 v16h;
typedef __attribute__((ext_vector_type(8)))  float    v8f;

#define NNODES 50000
#define NEDGES 1600000
#define NREL   4
#define FIN    32
#define HD1    32
#define HD2    64
#define NGRAPH 64
#define NCLS   3
#define KTOT   160   // (1 + NREL) * FIN

// ---------------- utility kernels ----------------

__global__ void k_zero(float* __restrict__ p, int n) {
    int i = blockIdx.x * blockDim.x + threadIdx.x;
    if (i < n) p[i] = 0.0f;
}

// per-relation in-degree counts (float so we can invert in place)
__global__ void k_count(const int* __restrict__ dst, const int* __restrict__ et,
                        float* __restrict__ cnt, int nE) {
    int e = blockIdx.x * blockDim.x + threadIdx.x;
    if (e >= nE) return;
    atomicAdd(&cnt[et[e] * NNODES + dst[e]], 1.0f);
}

__global__ void k_rcnt(float* __restrict__ c, int n) {
    int i = blockIdx.x * blockDim.x + threadIdx.x;
    if (i < n) c[i] = 1.0f / fmaxf(c[i], 1.0f);
}

// pack [root ; rel0..rel3] (each [FIN,Fout] row-major) into transposed f16 [Fout][KTOT]
__global__ void k_pack_w(const float* __restrict__ Wroot, const float* __restrict__ Wrel,
                         _Float16* __restrict__ WhT, int Fout) {
    int t = blockIdx.x * blockDim.x + threadIdx.x;
    if (t >= Fout * KTOT) return;
    int n = t / KTOT;
    int k = t - n * KTOT;
    float w = (k < FIN) ? Wroot[k * Fout + n]          // root block
                        : Wrel[(k - FIN) * Fout + n];  // (R,FIN,Fout) flat rows match k order
    WhT[n * KTOT + k] = (_Float16)w;
}

// sum raw source features per (relation, dst): agg[r][dst][0..31] += feats[src][0..31]
__global__ void k_edge_agg(const float* __restrict__ feats, const int* __restrict__ src,
                           const int* __restrict__ dst, const int* __restrict__ et,
                           float* __restrict__ agg, int nE) {
    int t = blockIdx.x * blockDim.x + threadIdx.x;
    int e = t >> 3;
    if (e >= nE) return;
    int c = (t & 7) * 4;
    int r = et[e];
    const float4 v = *(const float4*)(feats + src[e] * FIN + c);
    float* a = agg + ((r * NNODES + dst[e]) * FIN + c);
    atomicAdd(a + 0, v.x);
    atomicAdd(a + 1, v.y);
    atomicAdd(a + 2, v.z);
    atomicAdd(a + 3, v.w);
}

// ---------------- fused RGCN transform via WMMA ----------------
// Each wave computes a full 16 x FOUT row-block:
//   Y[node][col] = relu( bias[col] + sum_k u[node][k] * WhT[col][k] )
//   u = [ X | agg0*rc0 | agg1*rc1 | agg2*rc2 | agg3*rc3 ]  (K = 160, 5 wmma steps of 32)
// A operand is built once per K-step and reused across all FOUT/16 accumulators,
// amortizing the f32 loads + f32->f16 conversion across 2 (HD1) / 4 (HD2) WMMAs.
template <int FOUT>
__global__ void k_rgcn_gemm(const float* __restrict__ X, const float* __restrict__ agg,
                            const float* __restrict__ rcnt, const _Float16* __restrict__ WhT,
                            const float* __restrict__ bias, float* __restrict__ Y) {
    constexpr int TN = FOUT / 16;
    int wid  = (blockIdx.x * blockDim.x + threadIdx.x) >> 5;
    int lane = threadIdx.x & 31;
    if (wid >= NNODES / 16) return;       // wave-uniform guard: EXEC stays all-1s
    int hi = lane >> 4;       // half-wave select
    int l  = lane & 15;
    int node = wid * 16 + l;  // A-matrix row owned by this lane
    // A-operand K offsets inside each 32-wide segment (contiguous, matching B):
    // lanes 0-15: K in [0,16); lanes 16-31: K in [16,32)
    const int b1 = hi * 16;
    const int b2 = hi * 16 + 8;

    v8f acc[TN];
#pragma unroll
    for (int t = 0; t < TN; ++t) acc[t] = (v8f){};

#pragma unroll
    for (int s = 0; s < 5; ++s) {
        const float* rowp;
        float scale;
        if (s == 0) { rowp = X + node * FIN; scale = 1.0f; }
        else {
            int r = s - 1;
            rowp  = agg + (r * NNODES + node) * FIN;
            scale = rcnt[r * NNODES + node];      // mean = sum * (1/count)
        }
        float4 pa = *(const float4*)(rowp + b1);
        float4 pb = *(const float4*)(rowp + b1 + 4);
        float4 pc = *(const float4*)(rowp + b2);
        float4 pd = *(const float4*)(rowp + b2 + 4);
        v16h a;
        a[0]  = (_Float16)(pa.x * scale); a[1]  = (_Float16)(pa.y * scale);
        a[2]  = (_Float16)(pa.z * scale); a[3]  = (_Float16)(pa.w * scale);
        a[4]  = (_Float16)(pb.x * scale); a[5]  = (_Float16)(pb.y * scale);
        a[6]  = (_Float16)(pb.z * scale); a[7]  = (_Float16)(pb.w * scale);
        a[8]  = (_Float16)(pc.x * scale); a[9]  = (_Float16)(pc.y * scale);
        a[10] = (_Float16)(pc.z * scale); a[11] = (_Float16)(pc.w * scale);
        a[12] = (_Float16)(pd.x * scale); a[13] = (_Float16)(pd.y * scale);
        a[14] = (_Float16)(pd.z * scale); a[15] = (_Float16)(pd.w * scale);
#pragma unroll
        for (int t = 0; t < TN; ++t) {
            int col = t * 16 + l;  // B column owned by this lane for tile t
            // B: contiguous 16 halfs (lanes 0-15: K 0..15, lanes 16-31: K 16..31)
            v16h b = *(const v16h*)(WhT + col * KTOT + s * 32 + hi * 16);
            acc[t] = __builtin_amdgcn_wmma_f32_16x16x32_f16(false, a, false, b, (short)0,
                                                            acc[t], false, false);
        }
    }

    int rbase = wid * 16 + hi * 8;   // C/D layout: VGPR v -> row M = v + 8*hi
#pragma unroll
    for (int t = 0; t < TN; ++t) {
        int col = t * 16 + l;
        float bc = bias[col];
#pragma unroll
        for (int v = 0; v < 8; ++v) {
            float val = fmaxf(acc[t][v] + bc, 0.0f);
            Y[(rbase + v) * FOUT + col] = val;
        }
    }
}

// ---------------- pooling + head ----------------

// h >= 0 post-ReLU, so integer atomicMax on the float bits is order-correct; g init 0
__global__ void k_pool(const float* __restrict__ h, const int* __restrict__ batch,
                       float* __restrict__ g, int n) {
    int t = blockIdx.x * blockDim.x + threadIdx.x;
    if (t >= n) return;
    int node = t >> 6;            // HD2 == 64
    int f    = t & 63;
    atomicMax((int*)g + batch[node] * HD2 + f, __float_as_int(h[t]));
}

__global__ void k_head(const float* __restrict__ g, const float* __restrict__ w1,
                       const float* __restrict__ bb1, const float* __restrict__ w2,
                       const float* __restrict__ bb2, float* __restrict__ out) {
    int gi = threadIdx.x;
    if (gi >= NGRAPH) return;
    float hbuf[HD1];
#pragma unroll 4
    for (int j = 0; j < HD1; ++j) {
        float s = bb1[j];
        for (int k = 0; k < HD2; ++k) s += g[gi * HD2 + k] * w1[k * HD1 + j];
        hbuf[j] = fmaxf(s, 0.0f);
    }
    for (int cix = 0; cix < NCLS; ++cix) {
        float s = bb2[cix];
        for (int j = 0; j < HD1; ++j) s += hbuf[j] * w2[j * NCLS + cix];
        out[gi * NCLS + cix] = s;
    }
}

// ---------------- launcher ----------------

extern "C" void kernel_launch(void* const* d_in, const int* in_sizes, int n_in,
                              void* d_out, int out_size, void* d_ws, size_t ws_size,
                              hipStream_t stream) {
    (void)in_sizes; (void)n_in; (void)out_size; (void)ws_size;
    const float* x      = (const float*)d_in[0];
    const int*   eidx   = (const int*)d_in[1];
    const int*   srcp   = eidx;             // edge_index[0]
    const int*   dstp   = eidx + NEDGES;    // edge_index[1]
    const int*   et     = (const int*)d_in[2];
    const int*   batch  = (const int*)d_in[3];
    const float* W1rel  = (const float*)d_in[4];
    const float* W1root = (const float*)d_in[5];
    const float* b1     = (const float*)d_in[6];
    const float* W2rel  = (const float*)d_in[7];
    const float* W2root = (const float*)d_in[8];
    const float* b2     = (const float*)d_in[9];
    const float* l1w    = (const float*)d_in[10];
    const float* l1b    = (const float*)d_in[11];
    const float* l2w    = (const float*)d_in[12];
    const float* l2b    = (const float*)d_in[13];
    float* out = (float*)d_out;

    char* wsp = (char*)d_ws;
    auto alloc = [&](size_t bytes) {
        char* p = wsp;
        wsp += (bytes + 255) & ~(size_t)255;
        return p;
    };
    float*    agg  = (float*)alloc(sizeof(float) * NREL * NNODES * FIN);   // 25.6 MB
    float*    rcnt = (float*)alloc(sizeof(float) * NREL * NNODES);
    float*    h1   = (float*)alloc(sizeof(float) * NNODES * HD1);
    float*    h2   = (float*)alloc(sizeof(float) * NNODES * HD2);
    _Float16* WhT1 = (_Float16*)alloc(sizeof(_Float16) * HD1 * KTOT);
    _Float16* WhT2 = (_Float16*)alloc(sizeof(_Float16) * HD2 * KTOT);
    float*    gbuf = (float*)alloc(sizeof(float) * NGRAPH * HD2);

    const int B = 256;
    auto gr = [](int n, int b) { return (n + b - 1) / b; };

    // degree counts (shared by both layers) -> reciprocal in place
    k_zero <<<gr(NREL * NNODES, B), B, 0, stream>>>(rcnt, NREL * NNODES);
    k_count<<<gr(NEDGES, B),        B, 0, stream>>>(dstp, et, rcnt, NEDGES);
    k_rcnt <<<gr(NREL * NNODES, B), B, 0, stream>>>(rcnt, NREL * NNODES);

    // pack fused weights to transposed f16
    k_pack_w<<<gr(HD1 * KTOT, B), B, 0, stream>>>(W1root, W1rel, WhT1, HD1);
    k_pack_w<<<gr(HD2 * KTOT, B), B, 0, stream>>>(W2root, W2rel, WhT2, HD2);

    const int rowWaves = NNODES / 16;   // 3125 waves per transform

    // ---- layer 1: aggregate raw x, then fused WMMA transform + bias + relu ----
    k_zero    <<<gr(NREL * NNODES * FIN, B), B, 0, stream>>>(agg, NREL * NNODES * FIN);
    k_edge_agg<<<gr(NEDGES * 8, B),          B, 0, stream>>>(x, srcp, dstp, et, agg, NEDGES);
    k_rgcn_gemm<HD1><<<gr(rowWaves * 32, B), B, 0, stream>>>(x, agg, rcnt, WhT1, b1, h1);

    // ---- layer 2 ----
    k_zero    <<<gr(NREL * NNODES * FIN, B), B, 0, stream>>>(agg, NREL * NNODES * FIN);
    k_edge_agg<<<gr(NEDGES * 8, B),          B, 0, stream>>>(h1, srcp, dstp, et, agg, NEDGES);
    k_rgcn_gemm<HD2><<<gr(rowWaves * 32, B), B, 0, stream>>>(h1, agg, rcnt, WhT2, b2, h2);

    // ---- global max pool + MLP head ----
    k_zero<<<gr(NGRAPH * HD2, B), B, 0, stream>>>(gbuf, NGRAPH * HD2);
    k_pool<<<gr(NNODES * HD2, B), B, 0, stream>>>(h2, batch, gbuf, NNODES * HD2);
    k_head<<<1, 64, 0, stream>>>(gbuf, l1w, l1b, l2w, l2b, out);
}
